// MIGAttention_5059471475419
// MI455X (gfx1250) — matrix operand order repaired and checked by reading, not compiled, not run on hardware
//
#include <hip/hip_runtime.h>
#include <hip/hip_bf16.h>

typedef __attribute__((ext_vector_type(16))) _Float16 v16h;
typedef __attribute__((ext_vector_type(8)))  _Float16 v8h;
typedef __attribute__((ext_vector_type(4)))  _Float16 v4h;
typedef __attribute__((ext_vector_type(8)))  float    v8f;
typedef unsigned int __attribute__((ext_vector_type(4))) u32x4;
typedef int __attribute__((ext_vector_type(4)))          i32x4;
typedef int __attribute__((ext_vector_type(8)))          i32x8;

#if defined(__has_builtin)
#if __has_builtin(__builtin_amdgcn_tensor_load_to_lds) && \
    __has_builtin(__builtin_amdgcn_s_wait_tensorcnt)
#define USE_TDM 1
#endif
#endif
#ifndef USE_TDM
#define USE_TDM 0
#endif

// ---- WMMA fragment loaders (gfx1250 layouts, wave32) -----------------------
// A (16x32 f16): lane m=l&15; khb=(l>>4)*8; elems 0..7 = K khb..khb+7,
//                elems 8..15 = K khb+16..khb+23  -> two 16B loads at p, p+16h
__device__ __forceinline__ v16h ldA16(const _Float16* p) {
  v8h lo = *(const v8h*)p;
  v8h hi = *(const v8h*)(p + 16);
  v16h r;
#pragma unroll
  for (int i = 0; i < 8; ++i) { r[i] = lo[i]; r[i + 8] = hi[i]; }
  return r;
}
// B (32x16 f16) from Bt[n][k] row-major: lane n=l&15; khb=(l>>4)*8;
// elems 0..7 = K khb..khb+7, elems 8..15 = K khb+16..khb+23 -> two 16B loads at p, p+16h
__device__ __forceinline__ v16h ldB16(const _Float16* p) {
  v8h lo = *(const v8h*)p;
  v8h hi = *(const v8h*)(p + 16);
  v16h r;
#pragma unroll
  for (int i = 0; i < 8; ++i) { r[i] = lo[i]; r[i + 8] = hi[i]; }
  return r;
}

#if USE_TDM
// ---- Tensor Data Mover: 2-D tile (f16) global -> LDS with row padding ------
// D# per CDNA5 ISA: group0 = count/lds/global/type,
// group1 = flags/dims/strides.  pad_interval: 1<<pi 8-byte units per row,
// pad_amount: pa+1 dwords appended -> padded LDS row stride.
__device__ __forceinline__ void tdm_load_2d_f16(
    unsigned lds_off, const void* gaddr, unsigned tdim0, unsigned tdim1,
    unsigned long long stride0, unsigned tile0, unsigned tile1,
    unsigned pad_interval, unsigned pad_amount) {
  unsigned long long ga = (unsigned long long)(uintptr_t)gaddr;
  u32x4 g0;
  g0[0] = 1u;                                           // count=1, no gather
  g0[1] = lds_off;                                      // lds_addr
  g0[2] = (unsigned)ga;                                 // global_addr[31:0]
  g0[3] = (unsigned)((ga >> 32) & 0x01FFFFFFu) | (2u << 30);  // [56:32] | type=2
  i32x8 g1;
  g1[0] = (int)((1u << 16) |                            // data_size = 2B
                (1u << 20) |                            // pad_enable
                (pad_interval << 22) | (pad_amount << 25));
  g1[1] = (int)((tdim0 & 0xFFFFu) << 16);               // tensor_dim0[15:0]
  g1[2] = (int)((tdim0 >> 16) | ((tdim1 & 0xFFFFu) << 16));
  g1[3] = (int)((tdim1 >> 16) | (tile0 << 16));         // tile_dim0
  g1[4] = (int)tile1;                                   // tile_dim1 (tile_dim2=0)
  g1[5] = (int)(unsigned)stride0;                       // tensor_dim0_stride lo
  g1[6] = (int)((unsigned)(stride0 >> 32) & 0xFFFFu);   // stride hi | dim1_stride lo
  g1[7] = 0;
  i32x4 z4 = {0, 0, 0, 0};
#if __clang_major__ >= 23
  i32x8 z8 = {0, 0, 0, 0, 0, 0, 0, 0};
  __builtin_amdgcn_tensor_load_to_lds(g0, g1, z4, z4, z8, 0);
#else
  __builtin_amdgcn_tensor_load_to_lds(g0, g1, z4, z4, 0);
#endif
}
#endif

// ---- prep kernels ----------------------------------------------------------
__global__ void transpose_cast_kernel(const float* __restrict__ W,
                                      _Float16* __restrict__ Wt,
                                      int K, int N) {
  long o = (long)blockIdx.x * blockDim.x + threadIdx.x;
  if (o >= (long)K * N) return;
  int n = (int)(o / K), k = (int)(o % K);
  Wt[o] = (_Float16)W[(long)k * N + n];
}

__global__ void gate_kernel(const float* __restrict__ x,
                            const float* __restrict__ rw,
                            const float* __restrict__ rb,
                            float* __restrict__ gates, int C) {
  int lane = threadIdx.x & 31, wid = threadIdx.x >> 5;
  int tok = blockIdx.x * 8 + wid;
  const float* xp = x + (long)tok * C;
  float s = 0.f;
  for (int j = lane; j < C; j += 32) s += xp[j] * rw[j];
#pragma unroll
  for (int m = 16; m > 0; m >>= 1) s += __shfl_xor(s, m, 32);
  if (lane == 0) gates[tok] = 1.f / (1.f + __expf(-(s + rb[0])));
}

__global__ __launch_bounds__(1024) void topk_mask_kernel(
    const float* __restrict__ gates, float* __restrict__ gm, int N, int k) {
  __shared__ float s[2048];
  int b = blockIdx.x, tid = threadIdx.x;
  for (int i = tid; i < N; i += 1024) s[i] = gates[(long)b * N + i];
  __syncthreads();
  for (int size = 2; size <= N; size <<= 1) {
    for (int stride = size >> 1; stride > 0; stride >>= 1) {
      for (int i = tid; i < N; i += 1024) {
        int j = i ^ stride;
        if (j > i) {
          bool up = ((i & size) == 0);
          float a = s[i], c = s[j];
          if ((a > c) == up) { s[i] = c; s[j] = a; }
        }
      }
      __syncthreads();
    }
  }
  float thr = s[N - k];  // k-th largest
  for (int i = tid; i < N; i += 1024) {
    float g = gates[(long)b * N + i];
    gm[(long)b * N + i] = (g >= thr) ? g : 0.f;
  }
}

__global__ void cast_xg_kernel(const float* __restrict__ x,
                               const float* __restrict__ gm,
                               _Float16* __restrict__ xg, int C) {
  long i = (long)blockIdx.x * blockDim.x + threadIdx.x;
  float4 v = ((const float4*)x)[i];
  float g = gm[(i * 4) / C];
  v4h o;
  o[0] = (_Float16)(v.x * g); o[1] = (_Float16)(v.y * g);
  o[2] = (_Float16)(v.z * g); o[3] = (_Float16)(v.w * g);
  ((v4h*)xg)[i] = o;
}

// ---- WMMA GEMM:  D = A[M,K] x Bt[N,K]^T, scaled + bias ---------------------
// block 256 thr = 8 waves; tile 128x64; wave tile 32x32 (2x2 WMMA); KSTEP 32
// LDS row stride 48 halves (96B): matches TDM pad (16dw data + 8dw pad).
template <bool OUT_HALF>
__global__ __launch_bounds__(256) void gemm_wmma_kernel(
    const _Float16* __restrict__ A, const _Float16* __restrict__ Bt,
    const float* __restrict__ bias, float oscale, void* __restrict__ D,
    int M, int N, int K) {
  constexpr int BM = 128, BN = 64, KS = 32, STR = 48;
  __shared__ _Float16 As[BM * STR];
  __shared__ _Float16 Bs[BN * STR];

  const int tid = threadIdx.x;
  const int lane = tid & 31, wid = tid >> 5;
  const int wm = wid & 3, wn = wid >> 2;
  const int ml = lane & 15, hi = lane >> 4, khb = hi * 8;
  const int bm = blockIdx.x * BM, bn = blockIdx.y * BN;

#if USE_TDM
  const unsigned ldsA = (unsigned)(uintptr_t)&As[0];
  const unsigned ldsB = (unsigned)(uintptr_t)&Bs[0];
#endif

  v8f acc[2][2] = {};

  for (int kk = 0; kk < K; kk += KS) {
    __syncthreads();
#if USE_TDM
    if (tid < 32) {  // wave 0 drives the Tensor Data Mover
      tdm_load_2d_f16(ldsA, A + (long)bm * K + kk, K, M, K, KS, BM, 3, 7);
      tdm_load_2d_f16(ldsB, Bt + (long)bn * K + kk, K, N, K, KS, BN, 3, 7);
      __builtin_amdgcn_s_wait_tensorcnt(0);
    }
#else
    {  // A tile 128x32
      int r = tid >> 1, c = (tid & 1) * 16;
      const _Float16* src = A + (long)(bm + r) * K + kk + c;
      *(uint4*)&As[r * STR + c]     = *(const uint4*)src;
      *(uint4*)&As[r * STR + c + 8] = *(const uint4*)(src + 8);
    }
    {  // B tile 64x32
      int r = tid >> 2, c = (tid & 3) * 8;
      *(uint4*)&Bs[r * STR + c] =
          *(const uint4*)(Bt + (long)(bn + r) * K + kk + c);
    }
#endif
    __syncthreads();

    v16h af[2], bf[2];
#pragma unroll
    for (int i = 0; i < 2; ++i)
      af[i] = ldA16(&As[(wm * 32 + i * 16 + ml) * STR + khb]);
#pragma unroll
    for (int j = 0; j < 2; ++j)
      bf[j] = ldB16(&Bs[(wn * 32 + j * 16 + ml) * STR + khb]);
#pragma unroll
    for (int i = 0; i < 2; ++i)
#pragma unroll
      for (int j = 0; j < 2; ++j)
        acc[i][j] = __builtin_amdgcn_wmma_f32_16x16x32_f16(
            false, af[i], false, bf[j], (short)0, acc[i][j], false, false);
  }

#pragma unroll
  for (int i = 0; i < 2; ++i)
#pragma unroll
    for (int j = 0; j < 2; ++j) {
      int row0 = bm + wm * 32 + i * 16 + hi * 8;
      int col = bn + wn * 32 + j * 16 + ml;
      float bb = bias[col];
#pragma unroll
      for (int r = 0; r < 8; ++r) {
        float v = (acc[i][j][r] + bb) * oscale;
        long off = (long)(row0 + r) * N + col;
        if (OUT_HALF) ((_Float16*)D)[off] = (_Float16)v;
        else          ((float*)D)[off] = v;
      }
    }
}

// ---- flash attention (GQA, full softmax) ----------------------------------
// grid (N/64, B*H), block 128 (4 waves); wave owns 16 query rows.
// 64-key tiles: 16 WMMAs per iteration, softmax shuffles amortized 2x.
// Q was pre-scaled by 1/sqrt(DH) in the Q-projection epilogue.
__global__ __launch_bounds__(128) void attn_kernel(
    const _Float16* __restrict__ Qb, const _Float16* __restrict__ Kb,
    const _Float16* __restrict__ Vb, _Float16* __restrict__ O16) {
  constexpr int LDK = 72;  // Ks: 64 x (64+8)  144B rows == TDM pad (32+4 dw)
  constexpr int LDV = 80;  // Vt: 64 x (64+16) 160B rows
  constexpr int LDP = 80;  // Ps: 4*16 x (64+16)
  __shared__ _Float16 Ks[64 * LDK];
  __shared__ _Float16 Vt[64 * LDV];
  __shared__ _Float16 Ps[4 * 16 * LDP];

  const int tid = threadIdx.x;
  const int lane = tid & 31, w = tid >> 5;
  const int ml = lane & 15, hi = lane >> 4, khb = hi * 8;

  const int bh = blockIdx.y;
  const int b = bh >> 4, h = bh & 15, hk = h >> 2;  // H=16, rep=4
  const int qbase = blockIdx.x * 64 + w * 16;

  const _Float16* Qp = Qb + (long)b * 2048 * 1024 + (long)h * 64;
  const _Float16* Kp = Kb + (long)b * 2048 * 256 + (long)hk * 64;
  const _Float16* Vp = Vb + (long)b * 2048 * 256 + (long)hk * 64;

#if USE_TDM
  const unsigned ldsK = (unsigned)(uintptr_t)&Ks[0];
#endif

  v16h qa[2];
  {
    const _Float16* qr = Qp + (long)(qbase + ml) * 1024;
    qa[0] = ldA16(qr + khb);
    qa[1] = ldA16(qr + 32 + khb);
  }

  v8f o[4] = {};
  float m_i[8], l_i[8];
#pragma unroll
  for (int r = 0; r < 8; ++r) { m_i[r] = -1e30f; l_i[r] = 0.f; }

  for (int kt = 0; kt < 2048; kt += 64) {
    __syncthreads();
#if USE_TDM
    if (tid < 32)  // K tile 64x64 via TDM (pad 32dw rows by 4dw -> 72h stride)
      tdm_load_2d_f16(ldsK, Kp + (long)kt * 256, 256, 2048, 256, 64, 64, 4, 3);
#else
    {  // K tile 64x64 row-major (== B^t for QK^T)
      int r = tid >> 1, c = (tid & 1) * 32;
      const _Float16* src = Kp + (long)(kt + r) * 256 + c;
#pragma unroll
      for (int u = 0; u < 4; ++u)
        *(uint4*)&Ks[r * LDK + c + u * 8] = *(const uint4*)(src + u * 8);
    }
#endif
    {  // V tile 64x64 -> transposed Vt[dh][key] (== B^t for P.V)
      int r = tid >> 1, cc = (tid & 1) * 32;
      const _Float16* vp = Vp + (long)(kt + r) * 256 + cc;
#pragma unroll
      for (int q2 = 0; q2 < 32; ++q2) Vt[(cc + q2) * LDV + r] = vp[q2];
    }
    if (kt + 64 < 2048) {  // hint next tiles toward L2/WGP$
      __builtin_prefetch(Kp + (long)(kt + 64) * 256, 0, 0);
      __builtin_prefetch(Vp + (long)(kt + 64) * 256, 0, 0);
    }
#if USE_TDM
    if (tid < 32) __builtin_amdgcn_s_wait_tensorcnt(0);
#endif
    __syncthreads();

    // S = Q K^T (Q pre-scaled); 64 keys = 4 column tiles
    v8f sc[4] = {};
#pragma unroll
    for (int ss = 0; ss < 2; ++ss)
#pragma unroll
      for (int j = 0; j < 4; ++j) {
        v16h kb = ldB16(&Ks[(j * 16 + ml) * LDK + ss * 32 + khb]);
        sc[j] = __builtin_amdgcn_wmma_f32_16x16x32_f16(
            false, qa[ss], false, kb, (short)0, sc[j], false, false);
      }

    // online softmax; row r+8*hi lives across the 16 lanes of this half-wave
    float alpha[8];
#pragma unroll
    for (int r = 0; r < 8; ++r) {
      float v0 = sc[0][r], v1 = sc[1][r], v2 = sc[2][r], v3 = sc[3][r];
      float lm = fmaxf(fmaxf(v0, v1), fmaxf(v2, v3));
      lm = fmaxf(lm, __shfl_xor(lm, 1, 32));
      lm = fmaxf(lm, __shfl_xor(lm, 2, 32));
      lm = fmaxf(lm, __shfl_xor(lm, 4, 32));
      lm = fmaxf(lm, __shfl_xor(lm, 8, 32));
      float mn = fmaxf(m_i[r], lm);
      alpha[r] = __expf(m_i[r] - mn);
      m_i[r] = mn;
      float p0 = __expf(v0 - mn), p1 = __expf(v1 - mn);
      float p2 = __expf(v2 - mn), p3 = __expf(v3 - mn);
      float rs = (p0 + p1) + (p2 + p3);
      rs += __shfl_xor(rs, 1, 32);
      rs += __shfl_xor(rs, 2, 32);
      rs += __shfl_xor(rs, 4, 32);
      rs += __shfl_xor(rs, 8, 32);
      l_i[r] = l_i[r] * alpha[r] + rs;
      _Float16* pr = &Ps[(w * 16 + r + 8 * hi) * LDP];
      pr[ml]      = (_Float16)p0;
      pr[16 + ml] = (_Float16)p1;
      pr[32 + ml] = (_Float16)p2;
      pr[48 + ml] = (_Float16)p3;
    }
#pragma unroll
    for (int jd = 0; jd < 4; ++jd)
#pragma unroll
      for (int r = 0; r < 8; ++r) o[jd][r] *= alpha[r];
    __syncthreads();  // Ps C-layout -> A-layout round trip

    v16h pa[2];
#pragma unroll
    for (int s = 0; s < 2; ++s)
      pa[s] = ldA16(&Ps[(w * 16 + ml) * LDP + s * 32 + khb]);
#pragma unroll
    for (int jd = 0; jd < 4; ++jd)
#pragma unroll
      for (int s = 0; s < 2; ++s) {
        v16h vb = ldB16(&Vt[(jd * 16 + ml) * LDV + s * 32 + khb]);
        o[jd] = __builtin_amdgcn_wmma_f32_16x16x32_f16(
            false, pa[s], false, vb, (short)0, o[jd], false, false);
      }
  }

#pragma unroll
  for (int r = 0; r < 8; ++r) l_i[r] = 1.f / l_i[r];
#pragma unroll
  for (int jd = 0; jd < 4; ++jd)
#pragma unroll
    for (int r = 0; r < 8; ++r) {
      long row = (long)b * 2048 + qbase + r + 8 * hi;
      O16[row * 1024 + h * 64 + jd * 16 + ml] = (_Float16)(o[jd][r] * l_i[r]);
    }
}

// ---- host orchestration ----------------------------------------------------
extern "C" void kernel_launch(void* const* d_in, const int* in_sizes, int n_in,
                              void* d_out, int out_size, void* d_ws, size_t ws_size,
                              hipStream_t stream) {
  const float* x  = (const float*)d_in[0];
  const float* rw = (const float*)d_in[1];
  const float* rb = (const float*)d_in[2];
  const float* wq = (const float*)d_in[3];
  const float* bq = (const float*)d_in[4];
  const float* wk = (const float*)d_in[5];
  const float* bk = (const float*)d_in[6];
  const float* wv = (const float*)d_in[7];
  const float* bv = (const float*)d_in[8];
  const float* wo = (const float*)d_in[9];
  const float* bo = (const float*)d_in[10];
  float* out = (float*)d_out;

  const int B = 4, N = 2048, C = 1024;
  const int M = B * N;   // 8192 tokens
  const int KSEL = 1433; // int(2048*0.7)

  char* ws = (char*)d_ws;
  size_t off = 0;
  auto alloc = [&](size_t bytes) -> void* {
    void* p = ws + off;
    off += (bytes + 255) & ~(size_t)255;
    return p;
  };
  float*    gates = (float*)alloc((size_t)M * 4);
  float*    gm    = (float*)alloc((size_t)M * 4);
  _Float16* xg    = (_Float16*)alloc((size_t)M * C * 2);
  _Float16* wqt   = (_Float16*)alloc((size_t)C * 1024 * 2);
  _Float16* wkt   = (_Float16*)alloc((size_t)C * 256 * 2);
  _Float16* wvt   = (_Float16*)alloc((size_t)C * 256 * 2);
  _Float16* wot   = (_Float16*)alloc((size_t)C * 1024 * 2);
  _Float16* Qb    = (_Float16*)alloc((size_t)M * 1024 * 2);
  _Float16* Kb    = (_Float16*)alloc((size_t)M * 256 * 2);
  _Float16* Vb    = (_Float16*)alloc((size_t)M * 256 * 2);
  _Float16* Ob    = (_Float16*)alloc((size_t)M * 1024 * 2);

  transpose_cast_kernel<<<(C * 1024 + 255) / 256, 256, 0, stream>>>(wq, wqt, C, 1024);
  transpose_cast_kernel<<<(C * 256 + 255) / 256, 256, 0, stream>>>(wk, wkt, C, 256);
  transpose_cast_kernel<<<(C * 256 + 255) / 256, 256, 0, stream>>>(wv, wvt, C, 256);
  transpose_cast_kernel<<<(C * 1024 + 255) / 256, 256, 0, stream>>>(wo, wot, C, 1024);

  gate_kernel<<<M / 8, 256, 0, stream>>>(x, rw, rb, gates, C);
  topk_mask_kernel<<<B, 1024, 0, stream>>>(gates, gm, N, KSEL);
  cast_xg_kernel<<<(M * C / 4) / 256, 256, 0, stream>>>(x, gm, xg, C);

  // QKV projections (f16 out); Q pre-scaled by 1/sqrt(64)
  gemm_wmma_kernel<true><<<dim3(M / 128, 1024 / 64), 256, 0, stream>>>(xg, wqt, bq, 0.125f, Qb, M, 1024, C);
  gemm_wmma_kernel<true><<<dim3(M / 128, 256 / 64), 256, 0, stream>>>(xg, wkt, bk, 1.0f, Kb, M, 256, C);
  gemm_wmma_kernel<true><<<dim3(M / 128, 256 / 64), 256, 0, stream>>>(xg, wvt, bv, 1.0f, Vb, M, 256, C);

  attn_kernel<<<dim3(N / 64, B * 16), 128, 0, stream>>>(Qb, Kb, Vb, Ob);

  // output projection (f32 out)
  gemm_wmma_kernel<false><<<dim3(M / 128, C / 64), 256, 0, stream>>>(Ob, wot, bo, 1.0f, out, M, C, C);
}
